// FrameTokenLayer_54571854463204
// MI455X (gfx1250) — matrix operand order repaired and checked by reading, not compiled, not run on hardware
//
#include <hip/hip_runtime.h>
#include <hip/hip_bf16.h>

typedef unsigned short ushort_t;
typedef __attribute__((ext_vector_type(16))) __bf16 bf16x16;
typedef __attribute__((ext_vector_type(8)))  float  f32x8;
typedef __attribute__((ext_vector_type(4)))  float  f32x4;
typedef __attribute__((ext_vector_type(4)))  unsigned int u32x4;
typedef __attribute__((ext_vector_type(2)))  unsigned int u32x2;

__device__ __forceinline__ unsigned f2bf(float x) {
    union { float f; unsigned u; } v; v.f = x;
    unsigned r = v.u + 0x7FFFu + ((v.u >> 16) & 1u);   // round-to-nearest-even
    return r >> 16;
}
__device__ __forceinline__ unsigned pack2(float a, float b) {
    return f2bf(a) | (f2bf(b) << 16);
}

union FragBF { unsigned u[8]; bf16x16 v; };

// Async global -> LDS copy of 16 bytes (CDNA5, ASYNCcnt-tracked).
// Low 32 bits of the generic shared-pointer address are the LDS byte offset
// (ISA 10.2).
__device__ __forceinline__ void async_copy16(const void* gptr, void* lptr) {
    unsigned lds = (unsigned)(unsigned long long)lptr;
    asm volatile("global_load_async_to_lds_b128 %0, %1, off"
                 :: "v"(lds), "v"((unsigned long long)gptr) : "memory");
}
__device__ __forceinline__ void async_wait0() {
    asm volatile("s_wait_asynccnt 0x0" ::: "memory");
}
__device__ __forceinline__ void async_wait3() {
    asm volatile("s_wait_asynccnt 0x3" ::: "memory");
}

// ---------------------------------------------------------------------------
// GEMM: out = act(A[M,K]bf16 @ W[N,K]bf16^T + bias). Block tile 128x64,
// 8 waves, each wave owns a 32x32 register tile -> 4 v_wmma per K-step.
// Double-buffered LDS with ASYNCcnt-throttled software pipeline: the next
// K-panel's async loads are in flight while the current panel is multiplied.
// Optional f32 output and/or bf16 output (bf16 scaled by bscale).
// ---------------------------------------------------------------------------
__global__ __launch_bounds__(256) void gemm_bf16_kernel(
    const ushort_t* __restrict__ A, const ushort_t* __restrict__ W,
    const float* __restrict__ bias, float* __restrict__ outF,
    ushort_t* __restrict__ outB, int M, int N, int K, int relu, float bscale)
{
    __shared__ ushort_t As[2][128][32];
    __shared__ ushort_t Ws[2][64][32];

    const int tid  = threadIdx.x;
    const int lane = tid & 31;
    const int wave = tid >> 5;
    const int mw   = wave & 3;        // 4 m-waves * 32 rows = 128
    const int nw   = wave >> 2;       // 2 n-waves * 32 cols = 64
    const int m0   = blockIdx.y * 128;
    const int n0   = blockIdx.x * 64;
    const int l15  = lane & 15;
    const int kh   = lane >> 4;

    // loop-invariant staging coordinates (3 async 16B chunks per thread)
    const int ars0 = tid >> 2;           // A row, chunk 0 (rows 0..63)
    const int ars1 = (tid + 256) >> 2;   // A row, chunk 1 (rows 64..127)
    const int q8   = (tid & 3) << 3;     // 8-elem column offset
    int gra0 = m0 + ars0; if (gra0 >= M) gra0 = M - 1;
    int gra1 = m0 + ars1; if (gra1 >= M) gra1 = M - 1;
    const int grw = n0 + (tid >> 2);     // W row (N multiple of 64)

    f32x8 acc[2][2];
    acc[0][0] = (f32x8){}; acc[0][1] = (f32x8){};
    acc[1][0] = (f32x8){}; acc[1][1] = (f32x8){};

    auto issue = [&](int k0, int buf) {
        async_copy16(&A[(size_t)gra0 * K + k0 + q8], &As[buf][ars0][q8]);
        async_copy16(&A[(size_t)gra1 * K + k0 + q8], &As[buf][ars1][q8]);
        async_copy16(&W[(size_t)grw  * K + k0 + q8], &Ws[buf][tid >> 2][q8]);
    };

    const int nsteps = K >> 5;
    issue(0, 0);
    for (int s = 0; s < nsteps; ++s) {
        const int buf = s & 1;
        if (s + 1 < nsteps) {
            issue((s + 1) << 5, buf ^ 1);  // prefetch next panel (ping-pong)
            async_wait3();                  // current panel's 3 loads retired
        } else {
            async_wait0();
        }
        __syncthreads();

        FragBF fa0, fa1, fb0, fb1;
        const int ar0 = mw * 32 + l15, ar1 = ar0 + 16;
        const int br0 = nw * 32 + l15, br1 = br0 + 16;
#pragma unroll
        for (int t = 0; t < 8; ++t) {
            int ka = 2 * (t & 3) + 8 * kh + 16 * (t >> 2);  // A 16x32 layout
            fa0.u[t] = *(const unsigned*)&As[buf][ar0][ka];
            fa1.u[t] = *(const unsigned*)&As[buf][ar1][ka];
            int kb = 2 * (t & 3) + 8 * kh + 16 * (t >> 2);  // B 32x16 layout (same k order as A)
            fb0.u[t] = *(const unsigned*)&Ws[buf][br0][kb];
            fb1.u[t] = *(const unsigned*)&Ws[buf][br1][kb];
        }
        acc[0][0] = __builtin_amdgcn_wmma_f32_16x16x32_bf16(false, fa0.v, false, fb0.v, (short)0, acc[0][0], false, false);
        acc[0][1] = __builtin_amdgcn_wmma_f32_16x16x32_bf16(false, fa0.v, false, fb1.v, (short)0, acc[0][1], false, false);
        acc[1][0] = __builtin_amdgcn_wmma_f32_16x16x32_bf16(false, fa1.v, false, fb0.v, (short)0, acc[1][0], false, false);
        acc[1][1] = __builtin_amdgcn_wmma_f32_16x16x32_bf16(false, fa1.v, false, fb1.v, (short)0, acc[1][1], false, false);
        __syncthreads();   // all waves done with buf before it is refilled
    }

    // epilogue: D layout rows = r + 8*(lane>>4)
    const int rowb = m0 + mw * 32 + (kh << 3);
#pragma unroll
    for (int j = 0; j < 2; ++j) {
        const int ncol = n0 + nw * 32 + j * 16 + l15;
        const float bv = bias ? bias[ncol] : 0.f;
#pragma unroll
        for (int i = 0; i < 2; ++i) {
#pragma unroll
            for (int r = 0; r < 8; ++r) {
                int row = rowb + i * 16 + r;
                if (row < M) {
                    float vv = acc[i][j][r] + bv;
                    if (relu) vv = vv > 0.f ? vv : 0.f;
                    if (outF) outF[(size_t)row * N + ncol] = vv;
                    if (outB) outB[(size_t)row * N + ncol] = (ushort_t)f2bf(vv * bscale);
                }
            }
        }
    }
}

// ---------------------------------------------------------------------------
// Flash attention, bf16 in / bf16 out, f32 online softmax. H=8, Dh=32, C=256.
// Q already pre-scaled by Dh^-0.5 (folded into q-projection epilogue).
// grid.x = ceil(Lq/128); grid.y = B*H. Each wave = one 16-query tile.
// ---------------------------------------------------------------------------
__global__ __launch_bounds__(256) void attn_kernel(
    const ushort_t* __restrict__ Q, const ushort_t* __restrict__ K,
    const ushort_t* __restrict__ V, ushort_t* __restrict__ O,
    int Lq, int Lk, long long qStride, long long kStride)
{
    const int H = 8, C = 256;
    __shared__ ushort_t Ks[32][32];     // [key][dh]
    __shared__ ushort_t Vts[32][32];    // [dh][key]
    __shared__ ushort_t Qs[128][32];
    __shared__ ushort_t Ps[8][16][32];
    __shared__ float Sls[8][16][32];
    __shared__ float resc[8][16];

    const int tid  = threadIdx.x;
    const int lane = tid & 31;
    const int wave = tid >> 5;
    const int l15  = lane & 15;
    const int kh   = lane >> 4;
    const int b    = blockIdx.y / H, h = blockIdx.y % H;

    const ushort_t* qp = Q + (size_t)b * qStride + h * 32;
    const ushort_t* kp = K + (size_t)b * kStride + h * 32;
    const ushort_t* vp = V + (size_t)b * kStride + h * 32;
    ushort_t*       op = O + (size_t)b * qStride + h * 32;

    const int q0blk = blockIdx.x * 128;

    // stage 128 q rows x 32 dh (pure bf16 copy, 16B chunks)
#pragma unroll
    for (int cc = 0; cc < 2; ++cc) {
        int c = tid + cc * 256;
        int r = c >> 2, q8 = (c & 3) << 3;
        int gr = q0blk + r; if (gr >= Lq) gr = Lq - 1;
        *(u32x4*)&Qs[r][q8] = *(const u32x4*)&qp[(size_t)gr * C + q8];
    }
    __syncthreads();

    FragBF fq;
#pragma unroll
    for (int t = 0; t < 8; ++t) {
        int ka = 2 * (t & 3) + 8 * kh + 16 * (t >> 2);
        fq.u[t] = *(const unsigned*)&Qs[wave * 16 + l15][ka];
    }

    f32x8 o1 = {}, o2 = {};
    float mrun = -1e30f, lrun = 0.f;   // lanes 0..15: per-query-row stats
    const int nkb = (Lk + 31) / 32;

    for (int kb = 0; kb < nkb; ++kb) {
        const int kbase = kb * 32;
        __syncthreads();
        if (wave < 4) {                 // waves 0..3: K panel copy
            int r = tid >> 2, q8 = (tid & 3) << 3;
            int gk = kbase + r; if (gk >= Lk) gk = Lk - 1;
            *(u32x4*)&Ks[r][q8] = *(const u32x4*)&kp[(size_t)gk * C + q8];
        } else {                        // waves 4..7: V panel transpose
            int c = tid - 128;
            int r = c >> 2, q8 = (c & 3) << 3;
            int gk = kbase + r; if (gk >= Lk) gk = Lk - 1;
            union { u32x4 v; ushort_t hh[8]; } tv;
            tv.v = *(const u32x4*)&vp[(size_t)gk * C + q8];
#pragma unroll
            for (int j = 0; j < 8; ++j) Vts[q8 + j][r] = tv.hh[j];
        }
        __syncthreads();

        // S = Q @ K^T (two 16x16 tiles -> 32 keys)
        FragBF fk;
        f32x8 s1 = {}, s2 = {};
#pragma unroll
        for (int t = 0; t < 8; ++t) fk.u[t] = *(const unsigned*)&Ks[l15][2 * (t & 3) + 8 * kh + 16 * (t >> 2)];
        s1 = __builtin_amdgcn_wmma_f32_16x16x32_bf16(false, fq.v, false, fk.v, (short)0, s1, false, false);
#pragma unroll
        for (int t = 0; t < 8; ++t) fk.u[t] = *(const unsigned*)&Ks[16 + l15][2 * (t & 3) + 8 * kh + 16 * (t >> 2)];
        s2 = __builtin_amdgcn_wmma_f32_16x16x32_bf16(false, fq.v, false, fk.v, (short)0, s2, false, false);

        const int srow = kh << 3;
#pragma unroll
        for (int r = 0; r < 8; ++r) {
            Sls[wave][srow + r][l15]      = s1[r];
            Sls[wave][srow + r][16 + l15] = s2[r];
        }
        __syncthreads();

        if (lane < 16) {                // online softmax; lane = query row
            int nvalid = Lk - kbase; if (nvalid > 32) nvalid = 32;
            float mb = -1e30f;
            for (int j = 0; j < nvalid; ++j) {
                float sv = Sls[wave][lane][j];
                mb = sv > mb ? sv : mb;
            }
            float mnew = mrun > mb ? mrun : mb;
            float sc   = __expf(mrun - mnew);
            float ssum = 0.f;
            for (int j = 0; j < 32; ++j) {
                float e = (j < nvalid) ? __expf(Sls[wave][lane][j] - mnew) : 0.f;
                ssum += e;
                Ps[wave][lane][j] = (ushort_t)f2bf(e);
            }
            lrun = lrun * sc + ssum;
            mrun = mnew;
            resc[wave][lane] = sc;
        }
        __syncthreads();

        const int rowb = kh << 3;
#pragma unroll
        for (int r = 0; r < 8; ++r) {
            float f = resc[wave][rowb + r];
            o1[r] *= f; o2[r] *= f;
        }

        FragBF fp, fv;
#pragma unroll
        for (int t = 0; t < 8; ++t) {
            int ka = 2 * (t & 3) + 8 * kh + 16 * (t >> 2);
            fp.u[t] = *(const unsigned*)&Ps[wave][l15][ka];
        }
#pragma unroll
        for (int t = 0; t < 8; ++t) fv.u[t] = *(const unsigned*)&Vts[l15][2 * (t & 3) + 8 * kh + 16 * (t >> 2)];
        o1 = __builtin_amdgcn_wmma_f32_16x16x32_bf16(false, fp.v, false, fv.v, (short)0, o1, false, false);
#pragma unroll
        for (int t = 0; t < 8; ++t) fv.u[t] = *(const unsigned*)&Vts[16 + l15][2 * (t & 3) + 8 * kh + 16 * (t >> 2)];
        o2 = __builtin_amdgcn_wmma_f32_16x16x32_bf16(false, fp.v, false, fv.v, (short)0, o2, false, false);
    }

    __syncthreads();
    if (lane < 16) resc[wave][lane] = 1.f / lrun;
    __syncthreads();

    const int rowb = kh << 3;
#pragma unroll
    for (int r = 0; r < 8; ++r) {
        int grow = q0blk + wave * 16 + rowb + r;
        if (grow < Lq) {
            float inv = resc[wave][rowb + r];
            op[(size_t)grow * C + l15]      = (ushort_t)f2bf(o1[r] * inv);
            op[(size_t)grow * C + 16 + l15] = (ushort_t)f2bf(o2[r] * inv);
        }
    }
}

// ---------------------------------------------------------------------------
// f32 -> bf16 bulk convert (8 elems/thread)
// ---------------------------------------------------------------------------
__global__ void cvt_bf16_kernel(const float* __restrict__ a,
                                ushort_t* __restrict__ o, size_t n8) {
    size_t i = (size_t)blockIdx.x * blockDim.x + threadIdx.x;
    if (i >= n8) return;
    f32x4 a0 = ((const f32x4*)a)[2 * i];
    f32x4 a1 = ((const f32x4*)a)[2 * i + 1];
    u32x4 r;
    r.x = pack2(a0.x, a0.y); r.y = pack2(a0.z, a0.w);
    r.z = pack2(a1.x, a1.y); r.w = pack2(a1.z, a1.w);
    ((u32x4*)o)[i] = r;
}

// c_bf16 = a + b (4 elems/thread)
__global__ void add_bf16_kernel(const float* __restrict__ a, const float* __restrict__ b,
                                ushort_t* __restrict__ c, size_t n4) {
    size_t i = (size_t)blockIdx.x * blockDim.x + threadIdx.x;
    if (i >= n4) return;
    f32x4 av = ((const f32x4*)a)[i], bv = ((const f32x4*)b)[i];
    u32x2 r;
    r.x = pack2(av.x + bv.x, av.y + bv.y);
    r.y = pack2(av.z + bv.z, av.w + bv.w);
    ((u32x2*)c)[i] = r;
}

// ---------------------------------------------------------------------------
// Reference points: sigmoid(token @ ref_w^T + ref_b); one wave per row.
// ---------------------------------------------------------------------------
__global__ __launch_bounds__(256) void refpoint_kernel(
    const float* __restrict__ token, const float* __restrict__ rw,
    const float* __restrict__ rb, float* __restrict__ refp, int BT) {
    int wave = threadIdx.x >> 5, lane = threadIdx.x & 31;
    int i = blockIdx.x * 8 + wave;
    if (i >= BT) return;
    const float* t = token + (size_t)i * 256;
    float s0 = 0.f, s1 = 0.f;
    for (int j = lane; j < 256; j += 32) {
        float tv = t[j];
        s0 += tv * rw[j];
        s1 += tv * rw[256 + j];
    }
#pragma unroll
    for (int d = 16; d > 0; d >>= 1) {
        s0 += __shfl_xor(s0, d, 32);
        s1 += __shfl_xor(s1, d, 32);
    }
    if (lane == 0) {
        refp[i * 2 + 0] = 1.f / (1.f + __expf(-(s0 + rb[0])));
        refp[i * 2 + 1] = 1.f / (1.f + __expf(-(s1 + rb[1])));
    }
}

// softmax over 16 contiguous attention weights per (b,q,h); aw is [BT,128]
__global__ void softmax16_kernel(float* __restrict__ aw, int n16) {
    int i = blockIdx.x * blockDim.x + threadIdx.x;
    if (i >= n16) return;
    float* p = aw + (size_t)i * 16;
    float m = -1e30f;
    for (int j = 0; j < 16; ++j) m = p[j] > m ? p[j] : m;
    float s = 0.f;
    for (int j = 0; j < 16; ++j) { float e = __expf(p[j] - m); p[j] = e; s += e; }
    float inv = 1.f / s;
    for (int j = 0; j < 16; ++j) p[j] *= inv;
}

// ---------------------------------------------------------------------------
// Multi-scale deformable sampling. One wave per (b,q,h); lane = channel.
// Output written as bf16 (only consumer is the oproj GEMM).
// ---------------------------------------------------------------------------
__global__ __launch_bounds__(256) void msdeform_kernel(
    const float* __restrict__ value, const float* __restrict__ refp,
    const float* __restrict__ vr, const float* __restrict__ off,
    const float* __restrict__ aw, const int* __restrict__ ssh,
    const int* __restrict__ lsi, ushort_t* __restrict__ out)
{
    const int TO = 300, C = 256, Hn = 8, Ln = 4, Pn = 4, S = 5440;
    const int wave = threadIdx.x >> 5, lane = threadIdx.x & 31;
    const int gw = blockIdx.x * 8 + wave;
    const int h = gw % Hn;
    const int q = (gw / Hn) % TO;
    const int b = gw / (Hn * TO);
    if (b >= 8) return;

    const float rx = refp[(b * TO + q) * 2 + 0];
    const float ry = refp[(b * TO + q) * 2 + 1];
    const float* vbase = value + (size_t)b * S * C + h * 32 + lane;
    const float* offq  = off + (size_t)(b * TO + q) * 256;
    const float* awq   = aw  + (size_t)(b * TO + q) * 128;
    float acc = 0.f;

    for (int l = 0; l < Ln; ++l) {
        const int Hl = ssh[l * 2 + 0], Wl = ssh[l * 2 + 1];
        const int start = lsi[l];
        const float refx = rx * vr[(b * Ln + l) * 2 + 0];
        const float refy = ry * vr[(b * Ln + l) * 2 + 1];
        for (int p = 0; p < Pn; ++p) {
            const int nb = (h * Ln + l) * Pn + p;
            const float lx = (refx + offq[nb * 2 + 0] / (float)Wl) * Wl - 0.5f;
            const float ly = (refy + offq[nb * 2 + 1] / (float)Hl) * Hl - 0.5f;
            const float fx = floorf(lx), fy = floorf(ly);
            const int x0 = (int)fx, y0 = (int)fy;
            const float wx1 = lx - fx, wy1 = ly - fy;
            const float a = awq[nb];
            float samp = 0.f;
#pragma unroll
            for (int dx = 0; dx < 2; ++dx)
#pragma unroll
                for (int dy = 0; dy < 2; ++dy) {
                    int xi = x0 + dx, yi = y0 + dy;
                    float w = (dx ? wx1 : 1.f - wx1) * (dy ? wy1 : 1.f - wy1);
                    bool valid = (xi >= 0) && (xi < Wl) && (yi >= 0) && (yi < Hl);
                    int xc = xi < 0 ? 0 : (xi >= Wl ? Wl - 1 : xi);
                    int yc = yi < 0 ? 0 : (yi >= Hl ? Hl - 1 : yi);
                    float vv = vbase[(size_t)(start + yc * Wl + xc) * C];
                    samp += valid ? w * vv : 0.f;
                }
            acc += a * samp;
        }
    }
    out[(size_t)(b * TO + q) * 256 + h * 32 + lane] = (ushort_t)f2bf(acc);
}

// ---------------------------------------------------------------------------
// Residual-fused LayerNorm (C=256), optional bf16 mirror output.
// ---------------------------------------------------------------------------
__global__ __launch_bounds__(256) void ln_kernel(
    const float* __restrict__ x, const float* __restrict__ resid,
    const float* __restrict__ g, const float* __restrict__ beta,
    float* __restrict__ out, ushort_t* __restrict__ outB, int rows)
{
    const int C = 256;
    const int wave = threadIdx.x >> 5, lane = threadIdx.x & 31;
    const int row = blockIdx.x * 8 + wave;
    if (row >= rows) return;
    const float* xr = x + (size_t)row * C;
    const float* rr = resid + (size_t)row * C;
    float vals[8];
    float s = 0.f;
#pragma unroll
    for (int i = 0; i < 8; ++i) {
        float t = xr[lane + 32 * i] + rr[lane + 32 * i];
        vals[i] = t; s += t;
    }
#pragma unroll
    for (int d = 16; d > 0; d >>= 1) s += __shfl_xor(s, d, 32);
    const float mean = s * (1.f / 256.f);
    float vs = 0.f;
#pragma unroll
    for (int i = 0; i < 8; ++i) { float d = vals[i] - mean; vs += d * d; }
#pragma unroll
    for (int d = 16; d > 0; d >>= 1) vs += __shfl_xor(vs, d, 32);
    const float rstd = rsqrtf(vs * (1.f / 256.f) + 1e-5f);
#pragma unroll
    for (int i = 0; i < 8; ++i) {
        int c = lane + 32 * i;
        float o = (vals[i] - mean) * rstd * g[c] + beta[c];
        out[(size_t)row * C + c] = o;
        if (outB) outB[(size_t)row * C + c] = (ushort_t)f2bf(o);
    }
}

// ---------------------------------------------------------------------------
// Host orchestration
// ---------------------------------------------------------------------------
static inline void launch_gemm(const ushort_t* A, const ushort_t* W, const float* bias,
                               float* outF, ushort_t* outB, int M, int N, int K,
                               int relu, float bscale, hipStream_t s) {
    dim3 grid(N / 64, (M + 127) / 128);
    gemm_bf16_kernel<<<grid, 256, 0, s>>>(A, W, bias, outF, outB, M, N, K, relu, bscale);
}

extern "C" void kernel_launch(void* const* d_in, const int* in_sizes, int n_in,
                              void* d_out, int out_size, void* d_ws, size_t ws_size,
                              hipStream_t stream) {
    (void)in_sizes; (void)n_in; (void)out_size; (void)ws_size;
    const int B = 8, S = 5440, TO = 300, C = 256;
    const int BT = B * TO;        // 2400
    const int BS = B * S;         // 43520
    const float scale = 0.17677669529663687f;  // Dh^-0.5

    const float* src     = (const float*)d_in[0];
    const float* pos     = (const float*)d_in[1];
    const float* token   = (const float*)d_in[2];
    const float* tpose   = (const float*)d_in[3];
    const float* vr      = (const float*)d_in[4];
    const float* ref_w   = (const float*)d_in[5];
    const float* ref_b   = (const float*)d_in[6];
    const float* off_w   = (const float*)d_in[7];
    const float* off_b   = (const float*)d_in[8];
    const float* aw_w    = (const float*)d_in[9];
    const float* aw_b    = (const float*)d_in[10];
    const float* vproj_w = (const float*)d_in[11];
    const float* vproj_b = (const float*)d_in[12];
    const float* oproj_w = (const float*)d_in[13];
    const float* oproj_b = (const float*)d_in[14];
    const float* sa_in_w = (const float*)d_in[15];
    const float* sa_in_b = (const float*)d_in[16];
    const float* sa_out_w = (const float*)d_in[17];
    const float* sa_out_b = (const float*)d_in[18];
    const float* ca_in_w = (const float*)d_in[19];
    const float* ca_in_b = (const float*)d_in[20];
    const float* ca_out_w = (const float*)d_in[21];
    const float* ca_out_b = (const float*)d_in[22];
    const float* l1_w = (const float*)d_in[23];
    const float* l1_b = (const float*)d_in[24];
    const float* l2_w = (const float*)d_in[25];
    const float* l2_b = (const float*)d_in[26];
    const float* n1_g = (const float*)d_in[27];
    const float* n1_b = (const float*)d_in[28];
    const float* n2_g = (const float*)d_in[29];
    const float* n2_b = (const float*)d_in[30];
    const float* n3_g = (const float*)d_in[31];
    const float* n3_b = (const float*)d_in[32];
    const float* n4_g = (const float*)d_in[33];
    const float* n4_b = (const float*)d_in[34];
    const int*   ssh  = (const int*)d_in[35];
    const int*   lsi  = (const int*)d_in[36];

    // ---- workspace layout ----
    const size_t TC = (size_t)BT * C;          // 614400
    const size_t SC = (size_t)BS * C;          // 11141120
    float* wsf = (float*)d_ws;
    size_t o = 0;
    float* refp  = wsf + o; o += (size_t)BT * 2 + 64;
    float* offb  = wsf + o; o += TC;
    float* awb   = wsf + o; o += (size_t)BT * 128;
    float* t2    = wsf + o; o += TC;
    float* tok1  = wsf + o; o += TC;
    float* value = wsf + o; o += SC;
    float* s2b   = wsf + o; o += SC;
    float* src1  = wsf + o; o += SC;

    ushort_t* wsh = (ushort_t*)(wsf + o);
    size_t ho = 0;
    ushort_t* srcB   = wsh + ho; ho += SC;      // dead after vproj GEMM
    ushort_t* srcqB  = wsh + ho; ho += SC;      // dead after CA q GEMM
    ushort_t* qcaB   = wsh + ho; ho += SC;      // dead after CA attention
    ushort_t* aocaB  = wsh + ho; ho += SC;      // dead after ca_out GEMM
    ushort_t* ffnhB  = srcB;                    // overlay: 4*SC == BS*1024
    ushort_t* src1B  = wsh + ho; ho += SC;      // bf16 mirror of src1 (own slot)
    ushort_t* qdB    = wsh + ho; ho += TC;
    ushort_t* t2dB   = wsh + ho; ho += TC;
    ushort_t* tok1B  = wsh + ho; ho += TC;
    ushort_t* qkinB  = wsh + ho; ho += TC;
    ushort_t* qsaB   = wsh + ho; ho += TC;
    ushort_t* ksaB   = wsh + ho; ho += TC;
    ushort_t* vsaB   = wsh + ho; ho += TC;
    ushort_t* aosaB  = wsh + ho; ho += TC;
    ushort_t* tok2B  = wsh + ho; ho += TC;
    ushort_t* kincaB = wsh + ho; ho += TC;
    ushort_t* kcaB   = wsh + ho; ho += TC;
    ushort_t* vcaB   = wsh + ho; ho += TC;
    ushort_t* offwB  = wsh + ho; ho += 256 * 256;
    ushort_t* awwB   = wsh + ho; ho += 128 * 256;
    ushort_t* vprojB = wsh + ho; ho += 256 * 256;
    ushort_t* oprojB = wsh + ho; ho += 256 * 256;
    ushort_t* sainB  = wsh + ho; ho += 768 * 256;
    ushort_t* saoutB = wsh + ho; ho += 256 * 256;
    ushort_t* cainB  = wsh + ho; ho += 768 * 256;
    ushort_t* caoutB = wsh + ho; ho += 256 * 256;
    ushort_t* l1B    = wsh + ho; ho += 1024 * 256;
    ushort_t* l2B    = wsh + ho; ho += 256 * 1024;

    float* out_src = (float*)d_out;            // [BS, C]
    float* tok2    = (float*)d_out + SC;       // [BT, C] final token output

    auto cvt = [&](const float* a, ushort_t* ob, size_t n) {
        cvt_bf16_kernel<<<(unsigned)((n / 8 + 255) / 256), 256, 0, stream>>>(a, ob, n / 8);
    };
    auto addb = [&](const float* a, const float* b, ushort_t* c, size_t n) {
        add_bf16_kernel<<<(unsigned)((n / 4 + 255) / 256), 256, 0, stream>>>(a, b, c, n / 4);
    };

    // ---- one-time conversions (src + weights) ----
    cvt(src, srcB, SC);
    cvt(off_w, offwB, 256 * 256);    cvt(aw_w, awwB, 128 * 256);
    cvt(vproj_w, vprojB, 256 * 256); cvt(oproj_w, oprojB, 256 * 256);
    cvt(sa_in_w, sainB, 768 * 256);  cvt(sa_out_w, saoutB, 256 * 256);
    cvt(ca_in_w, cainB, 768 * 256);  cvt(ca_out_w, caoutB, 256 * 256);
    cvt(l1_w, l1B, 1024 * 256);      cvt(l2_w, l2B, 256 * 1024);

    // ---- 1. deformable cross-attention (token queries over src) ----
    addb(token, tpose, qdB, TC);
    refpoint_kernel<<<(BT + 7) / 8, 256, 0, stream>>>(token, ref_w, ref_b, refp, BT);
    launch_gemm(qdB, offwB, off_b, offb, nullptr, BT, 256, 256, 0, 1.f, stream);
    launch_gemm(qdB, awwB, aw_b, awb, nullptr, BT, 128, 256, 0, 1.f, stream);
    softmax16_kernel<<<(BT * 8 + 255) / 256, 256, 0, stream>>>(awb, BT * 8);
    launch_gemm(srcB, vprojB, vproj_b, value, nullptr, BS, 256, 256, 0, 1.f, stream);
    msdeform_kernel<<<(BT * 8) / 8, 256, 0, stream>>>(value, refp, vr, offb, awb, ssh, lsi, t2dB);
    launch_gemm(t2dB, oprojB, oproj_b, t2, nullptr, BT, 256, 256, 0, 1.f, stream);
    ln_kernel<<<(BT + 7) / 8, 256, 0, stream>>>(token, t2, n1_g, n1_b, tok1, tok1B, BT);

    // ---- 2. token self-attention (one 2400-long sequence) ----
    addb(tok1, tpose, qkinB, TC);
    launch_gemm(qkinB, sainB,             sa_in_b,       nullptr, qsaB, BT, 256, 256, 0, scale, stream);
    launch_gemm(qkinB, sainB + 256 * 256, sa_in_b + 256, nullptr, ksaB, BT, 256, 256, 0, 1.f, stream);
    launch_gemm(tok1B, sainB + 512 * 256, sa_in_b + 512, nullptr, vsaB, BT, 256, 256, 0, 1.f, stream);
    {
        dim3 grid((BT + 127) / 128, 8);  // batch=1, H=8
        attn_kernel<<<grid, 256, 0, stream>>>(qsaB, ksaB, vsaB, aosaB,
                                              BT, BT, (long long)BT * C, (long long)BT * C);
    }
    launch_gemm(aosaB, saoutB, sa_out_b, t2, nullptr, BT, 256, 256, 0, 1.f, stream);
    ln_kernel<<<(BT + 7) / 8, 256, 0, stream>>>(tok1, t2, n2_g, n2_b, tok2, tok2B, BT);

    // ---- 3. frame <- token cross-attention ----
    addb(src, pos, srcqB, SC);
    launch_gemm(srcqB, cainB, ca_in_b, nullptr, qcaB, BS, 256, 256, 0, scale, stream);
    addb(tok2, tpose, kincaB, TC);
    launch_gemm(kincaB, cainB + 256 * 256, ca_in_b + 256, nullptr, kcaB, BT, 256, 256, 0, 1.f, stream);
    launch_gemm(tok2B,  cainB + 512 * 256, ca_in_b + 512, nullptr, vcaB, BT, 256, 256, 0, 1.f, stream);
    {
        dim3 grid((S + 127) / 128, B * 8);  // per-batch, Lk=300 (masked)
        attn_kernel<<<grid, 256, 0, stream>>>(qcaB, kcaB, vcaB, aocaB,
                                              S, TO, (long long)S * C, (long long)TO * C);
    }
    launch_gemm(aocaB, caoutB, ca_out_b, s2b, nullptr, BS, 256, 256, 0, 1.f, stream);
    ln_kernel<<<(BS + 7) / 8, 256, 0, stream>>>(src, s2b, n3_g, n3_b, src1, src1B, BS);

    // ---- 4. FFN ----
    launch_gemm(src1B, l1B, l1_b, nullptr, ffnhB, BS, 1024, 256, 1, 1.f, stream);
    launch_gemm(ffnhB, l2B, l2_b, s2b, nullptr, BS, 256, 1024, 0, 1.f, stream);
    ln_kernel<<<(BS + 7) / 8, 256, 0, stream>>>(src1, s2b, n4_g, n4_b, out_src, nullptr, BS);
}
